// Gru2_33414845563403
// MI455X (gfx1250) — matrix operand order repaired and checked by reading, not compiled, not run on hardware
//
#include <hip/hip_runtime.h>
#include <hip/hip_bf16.h>

// ---------------------------------------------------------------------------
// Types for CDNA5 WMMA (wave32)
// ---------------------------------------------------------------------------
typedef __attribute__((ext_vector_type(16))) _Float16     v16h;
typedef __attribute__((ext_vector_type(8)))  _Float16     v8h;
typedef __attribute__((ext_vector_type(8)))  float        v8f;
typedef __attribute__((ext_vector_type(4)))  unsigned int ui32x4;
typedef __attribute__((ext_vector_type(8)))  int          i32x8;
typedef __attribute__((ext_vector_type(4)))  int          i32x4;

#define GRU_B 256
#define GRU_S 512
#define GRU_H 128
#define G3H   384   // 3*H

#if defined(__has_include)
#  if __has_include(<hip/amd_detail/amd_gfx1250_TDM.h>)
#    define TDM_6ARG 1
#  else
#    define TDM_6ARG 0
#  endif
#else
#  define TDM_6ARG 0
#endif

__device__ __forceinline__ v8f wmma_f16(v16h a, v16h b, v8f c) {
  // D = A(16x32 f16) * B(32x16 f16) + C(16x16 f32)
  return __builtin_amdgcn_wmma_f32_16x16x32_f16(
      /*neg_a=*/false, a, /*neg_b=*/false, b,
      /*c_mod=*/(short)0, c, /*reuse_a=*/false, /*reuse_b=*/false);
}

// A fragment: 16x32 f16 tile, row-major source, leading dim `ld` (halves).
// ISA 7.12.2: lanes 0-15 -> M=lane, K in {0..7,16..23}; lanes 16-31 -> M=lane-16,
// K in {8..15,24..31}: two contiguous 8-half (16B) runs per lane.
__device__ __forceinline__ v16h frag_a_16x32(const _Float16* tile, int ld) {
  const int lane = threadIdx.x & 31;
  const int r  = lane & 15;
  const int kh = lane >> 4;
  const _Float16* p = tile + r * ld + kh * 8;
  v8h lo = *reinterpret_cast<const v8h*>(p);
  v8h hi = *reinterpret_cast<const v8h*>(p + 16);
  v16h v;
#pragma unroll
  for (int i = 0; i < 8; ++i) { v[i] = lo[i]; v[i + 8] = hi[i]; }
  return v;
}

// B fragment: 32x16 f16 (KxN). Source is row-major weight [N, K]: column n of B
// is row n of W (contiguous). Lanes 0-15 hold K in {0..7,16..23}, lanes 16-31 K in {8..15,24..31}.
__device__ __forceinline__ v16h frag_b_32x16(const _Float16* wrow, int ld) {
  const int lane = threadIdx.x & 31;
  const int n  = lane & 15;
  const int kh = lane >> 4;
  const _Float16* p = wrow + n * ld + kh * 8;
  v8h lo = *reinterpret_cast<const v8h*>(p);
  v8h hi = *reinterpret_cast<const v8h*>(p + 16);
  v16h v;
#pragma unroll
  for (int i = 0; i < 8; ++i) { v[i] = lo[i]; v[i + 8] = hi[i]; }
  return v;
}

__device__ __forceinline__ float sigmoid_fast(float x) {
  return 1.0f / (1.0f + __expf(-x));
}
__device__ __forceinline__ float tanh_fast(float x) {
  float e = __expf(-2.0f * x);
  return (1.0f - e) / (1.0f + e);
}

// ---------------------------------------------------------------------------
// TDM: 1-D contiguous f32 block, global -> LDS (CDNA5 Tensor Data Mover).
// Descriptor per CDNA5 ISA: group0 {count=1, lds_addr,
// global_addr, type=2}; group1 {data_size=4B, tensor_dim0=tile_dim0=nelems,
// tensor_dim1=1}; groups 2/3 zero (tensor <= 2D).
// ---------------------------------------------------------------------------
__device__ __forceinline__ void tdm_load_f32_1d(unsigned lds_byte_addr,
                                                const void* gptr,
                                                unsigned nelems) {
  unsigned long long ga = (unsigned long long)(uintptr_t)gptr;
  ui32x4 g0;
  g0[0] = 1u;                                              // count=1, user desc
  g0[1] = lds_byte_addr;                                   // lds_addr
  g0[2] = (unsigned)ga;                                    // global_addr[31:0]
  g0[3] = (unsigned)((ga >> 32) & 0x1FFFFFFu) | (2u << 30); // ga[56:32] | type=2
  i32x8 g1;
  g1[0] = (int)(2u << 16);                                 // data_size = 4 bytes
  g1[1] = (int)((nelems & 0xFFFFu) << 16);                 // tensor_dim0[15:0]
  g1[2] = (int)(((nelems >> 16) & 0xFFFFu) | (1u << 16));  // td0[31:16] | td1=1
  g1[3] = (int)((nelems & 0xFFFFu) << 16);                 // tile_dim0
  g1[4] = 0;                                               // tile_dim1/2 unused
  g1[5] = (int)nelems;                                     // dim0 stride
  g1[6] = 0;
  g1[7] = 0;
  i32x4 z4 = {0, 0, 0, 0};
#if TDM_6ARG
  i32x8 z8 = {0, 0, 0, 0, 0, 0, 0, 0};
  __builtin_amdgcn_tensor_load_to_lds(g0, g1, z4, z4, z8, 0);
#else
  __builtin_amdgcn_tensor_load_to_lds(g0, g1, z4, z4, 0);
#endif
}

__device__ __forceinline__ unsigned lds_addr_of(const void* p) {
  // Low 32 bits of a flat LDS address are the LDS byte offset (aperture layout).
  return (unsigned)(uintptr_t)p;
}

// ---------------------------------------------------------------------------
// f32 -> f16 weight conversion
// ---------------------------------------------------------------------------
__global__ void cvt_f16_kernel(const float* __restrict__ src,
                               _Float16* __restrict__ dst, int n) {
  int i = blockIdx.x * blockDim.x + threadIdx.x;
  if (i < n) dst[i] = (_Float16)src[i];
}

// ---------------------------------------------------------------------------
// Embedding gather + concat -> xh [B*S, 128] f16
// ---------------------------------------------------------------------------
__global__ void embed_kernel(const int* __restrict__ X, const int* __restrict__ L,
                             const float* __restrict__ emb,
                             _Float16* __restrict__ xh) {
  const int bs = blockIdx.x;       // 0 .. B*S-1
  const int e  = threadIdx.x;      // 0 .. 127
  const int tok = (e < 64) ? X[bs] : L[bs];
  xh[(size_t)bs * 128 + e] = (_Float16)emb[tok * 64 + (e & 63)];
}

// ---------------------------------------------------------------------------
// Input projection GEMM: xp[d][s][b][g] = x[b*S+s,:] . w[d][g,:] + b_ih[d][g]
// x: [B*S, K] f16 row-major; w: [2][384][K] f16; xp: [2][S][B][384] f32
// Templated on K (fully unrolled); each block processes MT M-tiles so the
// weight (B) fragments are loaded once into registers and reused.
// grid (B*S/16/MT, 3, 2), block 256 (8 waves, one 16x16 N-tile each)
// ---------------------------------------------------------------------------
template <int K, int MT>
__global__ __launch_bounds__(256, 2)
void xproj_kernel(const _Float16* __restrict__ x,
                  const _Float16* __restrict__ w,
                  const float* __restrict__ bih,
                  float* __restrict__ xp) {
  constexpr int KC = K / 32;
  const int lane = threadIdx.x & 31;
  const int wave = threadIdx.x >> 5;
  const int nt   = blockIdx.y * 8 + wave;  // 0..23
  const int d    = blockIdx.z;

  // Hoisted weight fragments (register-resident across the M strip)
  const _Float16* wb = w + ((size_t)d * G3H + nt * 16) * K;
  v16h bfrag[KC];
#pragma unroll
  for (int kc = 0; kc < KC; ++kc) bfrag[kc] = frag_b_32x16(wb + kc * 32, K);

  const int n  = lane & 15;
  const int g  = nt * 16 + n;
  const int mh = lane >> 4;
  const float bias = bih[d * G3H + g];

  for (int mt = 0; mt < MT; ++mt) {
    const int mtile = blockIdx.x * MT + mt;
    const _Float16* xa = x + (size_t)mtile * 16 * K;
    v8f acc = {};
#pragma unroll
    for (int kc = 0; kc < KC; ++kc)
      acc = wmma_f16(frag_a_16x32(xa + kc * 32, K), bfrag[kc], acc);
#pragma unroll
    for (int i = 0; i < 8; ++i) {
      int mg = mtile * 16 + mh * 8 + i;   // global row = b*S + s
      int b  = mg >> 9;                   // / S (S=512)
      int s  = mg & 511;
      xp[(((size_t)d * GRU_S + s) * GRU_B + b) * G3H + g] = acc[i] + bias;
    }
  }
}

// ---------------------------------------------------------------------------
// Recurrent scan: one workgroup per (batch-tile of 16, direction).
// h tile [16,128] double-buffered in LDS (f16); w_hh fragments preloaded into
// registers (12 x v16h -> 12 wmma per wave per step); the per-step 24KB xp
// tile is DMA'd into double-buffered LDS by the Tensor Data Mover one step
// ahead (wave 0 issues, s_wait_tensorcnt + workgroup barrier synchronize).
// __launch_bounds__(256, 1): only 32 WGs exist in the whole dispatch (one per
// WGP), so trade occupancy for VGPRs and keep all 12 w_hh fragments register-
// resident (round-2 asm showed scratch spill/reload of bfrag in the hot loop).
// xp: [2][S][B][384] f32 ; whh: [2][384][128] f16 ; hout: [B][S][256] f16
// grid (16, 2), block 256
// ---------------------------------------------------------------------------
__global__ __launch_bounds__(256, 1)
void gru_scan_kernel(const float* __restrict__ xp,
                     const _Float16* __restrict__ whh,
                     const float* __restrict__ bhh,
                     _Float16* __restrict__ hout) {
  __shared__ __align__(16) _Float16 hbuf[2][16][GRU_H];   //  8 KB
  __shared__ __align__(16) float    xbuf[2][16 * G3H];    // 48 KB

  const int lane = threadIdx.x & 31;
  const int wave = threadIdx.x >> 5;  // 0..7
  const int bt   = blockIdx.x;        // batch tile
  const int d    = blockIdx.y;        // direction
  const int n    = lane & 15;
  const int mh   = lane >> 4;
  const int j    = wave * 16 + n;     // hidden index owned by this lane

  // Preload w_hh B fragments: gate gi -> N-tile (gi*8 + wave); 4 K-chunks.
  v16h bfrag[3][4];
  float bias[3];
#pragma unroll
  for (int gi = 0; gi < 3; ++gi) {
    const int nt = gi * 8 + wave;
#pragma unroll
    for (int kc = 0; kc < 4; ++kc)
      bfrag[gi][kc] =
          frag_b_32x16(whh + ((size_t)d * G3H + nt * 16) * GRU_H + kc * 32, GRU_H);
    bias[gi] = bhh[d * G3H + nt * 16 + n];
  }

  // xp base for this (direction, batch tile); step t -> time index s
  const float* xp0 = xp + (((size_t)d * GRU_S) * GRU_B + bt * 16) * G3H;
#define XP_STEP(t) (xp0 + (size_t)(d ? (GRU_S - 1 - (t)) : (t)) * GRU_B * G3H)

  // h0 = 0
  for (int i = threadIdx.x; i < 16 * GRU_H; i += 256)
    (&hbuf[0][0][0])[i] = (_Float16)0.0f;
  v8f hprev = {};

  // Prime the xp pipeline: DMA step 0 into xbuf[0]
  if (wave == 0) {
    tdm_load_f32_1d(lds_addr_of(&xbuf[0][0]), XP_STEP(0), 16 * G3H);
    __builtin_amdgcn_s_wait_tensorcnt(0);
  }
  __syncthreads();

  for (int t = 0; t < GRU_S; ++t) {
    const int s = d ? (GRU_S - 1 - t) : t;

    // A fragments from the current LDS h buffer
    v16h a[4];
#pragma unroll
    for (int kc = 0; kc < 4; ++kc)
      a[kc] = frag_a_16x32(&hbuf[t & 1][0][kc * 32], GRU_H);

    // hg = h @ w_hh^T  (three 16x16 tiles per wave: r, z, n gates, same j)
    v8f hg[3];
#pragma unroll
    for (int gi = 0; gi < 3; ++gi) {
      v8f acc = {};
#pragma unroll
      for (int kc = 0; kc < 4; ++kc)
        acc = wmma_f16(a[kc], bfrag[gi][kc], acc);
      hg[gi] = acc;
    }

    // Kick off DMA for the next step's xp tile (overlaps with gate math)
    if (wave == 0 && t + 1 < GRU_S)
      tdm_load_f32_1d(lds_addr_of(&xbuf[(t + 1) & 1][0]), XP_STEP(t + 1),
                      16 * G3H);

    const float* xrow = &xbuf[t & 1][0];
#pragma unroll
    for (int i = 0; i < 8; ++i) {
      const int m = mh * 8 + i;
      const float xr = xrow[m * G3H + j];
      const float xz = xrow[m * G3H + GRU_H + j];
      const float xn = xrow[m * G3H + 2 * GRU_H + j];
      const float hr = hg[0][i] + bias[0];
      const float hz = hg[1][i] + bias[1];
      const float hn = hg[2][i] + bias[2];
      const float r  = sigmoid_fast(xr + hr);
      const float z  = sigmoid_fast(xz + hz);
      const float nn = tanh_fast(xn + r * hn);
      const float h2 = (1.0f - z) * nn + z * hprev[i];
      hprev[i] = h2;
      const _Float16 hh = (_Float16)h2;
      hbuf[(t + 1) & 1][m][j] = hh;                                   // next step A
      hout[(((size_t)(bt * 16 + m)) * GRU_S + s) * 256 + d * GRU_H + j] = hh;
    }

    // Ensure the next step's xp DMA has landed, then synchronize the WG
    if (wave == 0) __builtin_amdgcn_s_wait_tensorcnt(0);
    __syncthreads();
  }
#undef XP_STEP
}

// ---------------------------------------------------------------------------
// Mean pool over S: pooled[b][c] = mean_s h1[b][s][c]
// ---------------------------------------------------------------------------
__global__ void pool_kernel(const _Float16* __restrict__ h1,
                            float* __restrict__ pooled) {
  const int b = blockIdx.x, c = threadIdx.x;
  float acc = 0.0f;
  for (int s = 0; s < GRU_S; ++s)
    acc += (float)h1[((size_t)b * GRU_S + s) * 256 + c];
  pooled[b * 256 + c] = acc * (1.0f / (float)GRU_S);
}

// ---------------------------------------------------------------------------
// FC head: out = relu(pooled @ fc1^T + b1) @ fc2^T + b2
// ---------------------------------------------------------------------------
__global__ void head_kernel(const float* __restrict__ pooled,
                            const float* __restrict__ w1, const float* __restrict__ b1,
                            const float* __restrict__ w2, const float* __restrict__ b2,
                            float* __restrict__ out) {
  __shared__ float p[256];
  __shared__ float z[256];
  const int b = blockIdx.x, tid = threadIdx.x;
  p[tid] = pooled[b * 256 + tid];
  __syncthreads();
  float acc = b1[tid];
  for (int k = 0; k < 256; ++k) acc += p[k] * w1[tid * 256 + k];
  z[tid] = fmaxf(acc, 0.0f);
  __syncthreads();
  if (tid < 6) {
    float a2 = b2[tid];
    for (int k = 0; k < 256; ++k) a2 += z[k] * w2[tid * 256 + k];
    out[b * 6 + tid] = a2;
  }
}

// ---------------------------------------------------------------------------
// Launch
// ---------------------------------------------------------------------------
extern "C" void kernel_launch(void* const* d_in, const int* in_sizes, int n_in,
                              void* d_out, int out_size, void* d_ws, size_t ws_size,
                              hipStream_t stream) {
  const int*   X      = (const int*)  d_in[0];
  const int*   lemma  = (const int*)  d_in[1];
  const float* emb    = (const float*)d_in[2];
  const float* w_ih0  = (const float*)d_in[3];
  const float* w_hh0  = (const float*)d_in[4];
  const float* b_ih0  = (const float*)d_in[5];
  const float* b_hh0  = (const float*)d_in[6];
  const float* w_ih1  = (const float*)d_in[7];
  const float* w_hh1  = (const float*)d_in[8];
  const float* b_ih1  = (const float*)d_in[9];
  const float* b_hh1  = (const float*)d_in[10];
  const float* fc1_w  = (const float*)d_in[11];
  const float* fc1_b  = (const float*)d_in[12];
  const float* fc2_w  = (const float*)d_in[13];
  const float* fc2_b  = (const float*)d_in[14];
  float* out = (float*)d_out;

  // Workspace layout (256B aligned)
  char* ws = (char*)d_ws;
  size_t off = 0;
  auto alloc = [&](size_t bytes) {
    char* p = ws + off;
    off += (bytes + 255) & ~(size_t)255;
    return p;
  };
  _Float16* xh0    = (_Float16*)alloc((size_t)GRU_B * GRU_S * 128 * 2);     //  33.6 MB
  float*    xp     = (float*)   alloc((size_t)2 * GRU_S * GRU_B * G3H * 4); // 402.7 MB
  _Float16* h0     = (_Float16*)alloc((size_t)GRU_B * GRU_S * 256 * 2);     //  67.1 MB
  _Float16* h1     = (_Float16*)alloc((size_t)GRU_B * GRU_S * 256 * 2);     //  67.1 MB
  _Float16* wih0h  = (_Float16*)alloc((size_t)2 * G3H * 128 * 2);
  _Float16* whh0h  = (_Float16*)alloc((size_t)2 * G3H * 128 * 2);
  _Float16* wih1h  = (_Float16*)alloc((size_t)2 * G3H * 256 * 2);
  _Float16* whh1h  = (_Float16*)alloc((size_t)2 * G3H * 128 * 2);
  float*    pooled = (float*)   alloc((size_t)GRU_B * 256 * 4);

  // 1) Weights -> f16
  const int n_ih0 = 2 * G3H * 128, n_hh0 = 2 * G3H * 128;
  const int n_ih1 = 2 * G3H * 256, n_hh1 = 2 * G3H * 128;
  cvt_f16_kernel<<<(n_ih0 + 255) / 256, 256, 0, stream>>>(w_ih0, wih0h, n_ih0);
  cvt_f16_kernel<<<(n_hh0 + 255) / 256, 256, 0, stream>>>(w_hh0, whh0h, n_hh0);
  cvt_f16_kernel<<<(n_ih1 + 255) / 256, 256, 0, stream>>>(w_ih1, wih1h, n_ih1);
  cvt_f16_kernel<<<(n_hh1 + 255) / 256, 256, 0, stream>>>(w_hh1, whh1h, n_hh1);

  // 2) Embedding gather + concat (f16)
  embed_kernel<<<GRU_B * GRU_S, 128, 0, stream>>>(X, lemma, emb, xh0);

  // 3) Layer 0: input projection (WMMA GEMM), then recurrent scan (WMMA + TDM)
  constexpr int MT = 8;
  xproj_kernel<128, MT><<<dim3(GRU_B * GRU_S / 16 / MT, 3, 2), 256, 0, stream>>>(
      xh0, wih0h, b_ih0, xp);
  gru_scan_kernel<<<dim3(GRU_B / 16, 2), 256, 0, stream>>>(xp, whh0h, b_hh0, h0);

  // 4) Layer 1 (input = layer-0 output, K = 256); xp buffer reused
  xproj_kernel<256, MT><<<dim3(GRU_B * GRU_S / 16 / MT, 3, 2), 256, 0, stream>>>(
      h0, wih1h, b_ih1, xp);
  gru_scan_kernel<<<dim3(GRU_B / 16, 2), 256, 0, stream>>>(xp, whh1h, b_hh1, h1);

  // 5) Mean pool + FC head
  pool_kernel<<<GRU_B, 256, 0, stream>>>(h1, pooled);
  head_kernel<<<GRU_B, 256, 0, stream>>>(pooled, fc1_w, fc1_b, fc2_w, fc2_b, out);
}
